// DeformableTransformer_67826123538672
// MI455X (gfx1250) — matrix operand order repaired and checked by reading, not compiled, not run on hardware
//
#include <hip/hip_runtime.h>

// ---------------------------------------------------------------------------
// Deformable transformer decoder layer for MI455X (gfx1250, wave32, WMMA).
// All GEMMs run through V_WMMA_F32_16X16X4_F32 (exact fp32; workload is
// HBM-bound at 23.3 TB/s so fp32 WMMA throughput is not the limiter).
// Fallback: codegen-confirmed V_WMMA_F32_16X16X32_BF16 with f32 accumulate.
// ---------------------------------------------------------------------------

typedef __attribute__((ext_vector_type(2)))  float  v2f;
typedef __attribute__((ext_vector_type(8)))  float  v8f;
typedef __attribute__((ext_vector_type(16))) __bf16 v16bf;

#if defined(__AMDGCN__) && __has_builtin(__builtin_amdgcn_wmma_f32_16x16x4_f32)
#define USE_F32_WMMA 1
#endif

#define BS   8
#define LQ   300
#define DM   256
#define NHD  8
#define HD   32
#define NLV  4
#define NPT  4
#define DFF  1024
#define LV   20197
#define MQ   (BS * LQ)         // 2400
#define MV   (BS * LV)         // 161576

// ---------------------------------------------------------------------------
// GEMM: C[M,N] = A[M,K] @ W[N,K]^T + bias[N]   (optional ReLU / row mask)
// One wave -> 16x64 strip (4 accumulators reuse each A fragment 4x).
// blockDim = (32,2): block covers 16x128. K, N compile-time -> all address
// math folds to shifts and immediate offsets.
// ---------------------------------------------------------------------------
template <int K, int N, bool RELU, bool MASK, bool MGUARD>
__global__ __launch_bounds__(64) void gemm_wmma(
    const float* __restrict__ A, const float* __restrict__ W,
    const float* __restrict__ bias, float* __restrict__ C,
    int M, const unsigned char* __restrict__ rowmask) {
  const int lane = threadIdx.x;          // 0..31
  const int half = lane >> 4;            // 0 | 1
  const int lm   = lane & 15;
  const int m0   = blockIdx.x * 16;
  const int n0   = (blockIdx.y * 2 + threadIdx.y) * 64;

  int arow = m0 + lm;
  if (MGUARD && arow >= M) arow = M - 1;   // clamp; stores guarded below
  const float* Ap = A + (size_t)arow * K;
  const float* Wp = W + (size_t)(n0 + lm) * K;

  v8f acc0 = {}, acc1 = {}, acc2 = {}, acc3 = {};
#if defined(USE_F32_WMMA)
  // A 16x4 frag: lanes 0-15 -> K=k,k+1 ; lanes 16-31 -> K=k+2,k+3 (M = lane%16)
  // B 4x16 frag: same K split, N = lane%16. Contiguous float2 loads; the four
  // B tiles sit 16*K floats apart (compile-time immediate offsets).
#pragma unroll 4
  for (int k = 0; k < K; k += 4) {
    const int ko = k + 2 * half;
    v2f a  = *(const v2f*)(Ap + ko);
    v2f b0 = *(const v2f*)(Wp + ko);
    v2f b1 = *(const v2f*)(Wp + 16 * K + ko);
    v2f b2 = *(const v2f*)(Wp + 32 * K + ko);
    v2f b3 = *(const v2f*)(Wp + 48 * K + ko);
    acc0 = __builtin_amdgcn_wmma_f32_16x16x4_f32(false, a, false, b0, (short)0, acc0, false, false);
    acc1 = __builtin_amdgcn_wmma_f32_16x16x4_f32(false, a, false, b1, (short)0, acc1, false, false);
    acc2 = __builtin_amdgcn_wmma_f32_16x16x4_f32(false, a, false, b2, (short)0, acc2, false, false);
    acc3 = __builtin_amdgcn_wmma_f32_16x16x4_f32(false, a, false, b3, (short)0, acc3, false, false);
  }
#else
  // bf16 16x16x32 fallback (f32 accumulate). A layout: lanes 0-15 hold
  // K=k..k+15; lanes 16-31 hold K=k+16..k+31 (same order as B).
  // B layout: lanes 0-15 hold K=k..k+15, lanes 16-31 hold K=k+16..k+31.
  for (int k = 0; k < K; k += 32) {
    v16bf a;
    const float* A1 = Ap + k + (half ? 16 : 0);
    const float* A2 = Ap + k + (half ? 24 : 8);
#pragma unroll
    for (int e = 0; e < 8; ++e) { a[e] = (__bf16)A1[e]; a[8 + e] = (__bf16)A2[e]; }
    const float* B1 = Wp + k + 16 * half;
    v16bf b0, b1, b2, b3;
#pragma unroll
    for (int e = 0; e < 16; ++e) {
      b0[e] = (__bf16)B1[e];
      b1[e] = (__bf16)B1[16 * K + e];
      b2[e] = (__bf16)B1[32 * K + e];
      b3[e] = (__bf16)B1[48 * K + e];
    }
    acc0 = __builtin_amdgcn_wmma_f32_16x16x32_bf16(false, a, false, b0, (short)0, acc0, false, false);
    acc1 = __builtin_amdgcn_wmma_f32_16x16x32_bf16(false, a, false, b1, (short)0, acc1, false, false);
    acc2 = __builtin_amdgcn_wmma_f32_16x16x32_bf16(false, a, false, b2, (short)0, acc2, false, false);
    acc3 = __builtin_amdgcn_wmma_f32_16x16x32_bf16(false, a, false, b3, (short)0, acc3, false, false);
  }
#endif
  // C/D layout: VGPR r -> M = r (lanes 0-15) | r+8 (lanes 16-31), N = lane%16.
  const int n = n0 + lm;
  const float b0 = bias[n], b1 = bias[n + 16], b2 = bias[n + 32], b3 = bias[n + 48];
#pragma unroll
  for (int r = 0; r < 8; ++r) {
    const int m = m0 + r + 8 * half;
    if (!MGUARD || m < M) {
      float v0 = acc0[r] + b0, v1 = acc1[r] + b1, v2 = acc2[r] + b2, v3 = acc3[r] + b3;
      if (RELU) {
        v0 = fmaxf(v0, 0.0f); v1 = fmaxf(v1, 0.0f);
        v2 = fmaxf(v2, 0.0f); v3 = fmaxf(v3, 0.0f);
      }
      if (MASK && rowmask[m]) { v0 = v1 = v2 = v3 = 0.0f; }
      float* crow = C + (size_t)m * N + n;
      crow[0] = v0; crow[16] = v1; crow[32] = v2; crow[48] = v3;
    }
  }
}

// ---------------------------------------------------------------------------
// Elementwise add
// ---------------------------------------------------------------------------
__global__ void add_kernel(const float* __restrict__ a, const float* __restrict__ b,
                           float* __restrict__ o, int n) {
  int i = blockIdx.x * blockDim.x + threadIdx.x;
  if (i < n) o[i] = a[i] + b[i];
}

// ---------------------------------------------------------------------------
// out = LayerNorm(a + b) over 256 channels; one wave per row.
// ---------------------------------------------------------------------------
__global__ __launch_bounds__(256) void add_ln_kernel(
    const float* __restrict__ a, const float* __restrict__ b,
    const float* __restrict__ w, const float* __restrict__ bias,
    float* __restrict__ out, int rows) {
  const int wid  = (blockIdx.x * blockDim.x + threadIdx.x) >> 5;
  const int lane = threadIdx.x & 31;
  if (wid >= rows) return;
  const float* ar = a + (size_t)wid * DM;
  const float* br = b + (size_t)wid * DM;
  float v[8], s = 0.0f;
#pragma unroll
  for (int i = 0; i < 8; ++i) { v[i] = ar[lane + 32 * i] + br[lane + 32 * i]; s += v[i]; }
#pragma unroll
  for (int off = 16; off; off >>= 1) s += __shfl_xor(s, off, 32);
  const float mean = s * (1.0f / DM);
  float s2 = 0.0f;
#pragma unroll
  for (int i = 0; i < 8; ++i) { const float d = v[i] - mean; s2 += d * d; }
#pragma unroll
  for (int off = 16; off; off >>= 1) s2 += __shfl_xor(s2, off, 32);
  const float rstd = rsqrtf(s2 * (1.0f / DM) + 1e-5f);
  float* orow = out + (size_t)wid * DM;
#pragma unroll
  for (int i = 0; i < 8; ++i) {
    const int c = lane + 32 * i;
    orow[c] = (v[i] - mean) * rstd * w[c] + bias[c];
  }
}

// ---------------------------------------------------------------------------
// Self-attention: one wave per (b,h,q). qkbuf rows: [0:256]=q, [256:512]=k.
// ---------------------------------------------------------------------------
__global__ __launch_bounds__(32) void self_attn_kernel(
    const float* __restrict__ qkbuf, const float* __restrict__ vbuf,
    float* __restrict__ out) {
  const int idx = blockIdx.x;               // b*NHD*LQ + h*LQ + q
  const int q = idx % LQ;
  const int h = (idx / LQ) % NHD;
  const int b = idx / (LQ * NHD);
  const int lane = threadIdx.x;
  __shared__ float sp[LQ + 4];

  const float* qrow = qkbuf + (size_t)(b * LQ + q) * 512 + h * HD;
  const float scale = 0.1767766952966369f;  // 1/sqrt(32)

  float lmax = -1e30f;
  for (int k = lane; k < LQ; k += 32) {
    const float* krow = qkbuf + (size_t)(b * LQ + k) * 512 + DM + h * HD;
    float s = 0.0f;
#pragma unroll
    for (int d = 0; d < HD; ++d) s += qrow[d] * krow[d];
    s *= scale;
    sp[k] = s;
    lmax = fmaxf(lmax, s);
  }
#pragma unroll
  for (int off = 16; off; off >>= 1) lmax = fmaxf(lmax, __shfl_xor(lmax, off, 32));
  __syncthreads();
  float lsum = 0.0f;
  for (int k = lane; k < LQ; k += 32) {
    const float e = __expf(sp[k] - lmax);
    sp[k] = e;
    lsum += e;
  }
#pragma unroll
  for (int off = 16; off; off >>= 1) lsum += __shfl_xor(lsum, off, 32);
  __syncthreads();
  const float inv = 1.0f / lsum;
  float acc = 0.0f;
  for (int k = 0; k < LQ; ++k)
    acc += sp[k] * vbuf[(size_t)(b * LQ + k) * DM + h * HD + lane];
  out[(size_t)(b * LQ + q) * DM + h * HD + lane] = acc * inv;
}

// ---------------------------------------------------------------------------
// Sampling-location + attention-weight softmax. One thread per (b,q,h).
// Writes loc (b,q,h,l,p,2) and aw (b,q,h,l,p) output tensors directly.
// ---------------------------------------------------------------------------
__global__ void loc_aw_kernel(const float* __restrict__ logits,
                              const float* __restrict__ offsets,
                              const float* __restrict__ refp,
                              float* __restrict__ loc_out,
                              float* __restrict__ aw_out) {
  const int t = blockIdx.x * blockDim.x + threadIdx.x;
  if (t >= MQ * NHD) return;
  const int h  = t & 7;
  const int bq = t >> 3;

  const float* lg = logits + (size_t)bq * 128 + h * 16;
  float m = -1e30f;
#pragma unroll
  for (int j = 0; j < 16; ++j) m = fmaxf(m, lg[j]);
  float e[16], s = 0.0f;
#pragma unroll
  for (int j = 0; j < 16; ++j) { e[j] = __expf(lg[j] - m); s += e[j]; }
  const float inv = 1.0f / s;
  float* awp = aw_out + (size_t)t * 16;
#pragma unroll
  for (int j = 0; j < 16; ++j) awp[j] = e[j] * inv;

  const float LWf[4] = {152.0f, 76.0f, 38.0f, 19.0f};
  const float LHf[4] = {100.0f, 50.0f, 25.0f, 13.0f};
  const float* off = offsets + (size_t)bq * 256 + h * 32;  // (l*4+p)*2+c
  float* lp = loc_out + (size_t)t * 32;
#pragma unroll
  for (int l = 0; l < NLV; ++l) {
    const float rx = refp[((size_t)bq * NLV + l) * 2 + 0];
    const float ry = refp[((size_t)bq * NLV + l) * 2 + 1];
    const float iw = 1.0f / LWf[l], ih = 1.0f / LHf[l];
#pragma unroll
    for (int p = 0; p < NPT; ++p) {
      const int j = l * 4 + p;
      lp[j * 2 + 0] = rx + off[j * 2 + 0] * iw;
      lp[j * 2 + 1] = ry + off[j * 2 + 1] * ih;
    }
  }
}

// ---------------------------------------------------------------------------
// Deformable bilinear sampling. One wave per (b,q,h); lane = head channel.
// value layout: (b, flat_idx, h, d) with flat stride NHD*HD = 256.
// ---------------------------------------------------------------------------
__global__ __launch_bounds__(256) void deform_sample_kernel(
    const float* __restrict__ value, const float* __restrict__ loc,
    const float* __restrict__ aw, float* __restrict__ out) {
  const int wid  = (blockIdx.x * blockDim.x + threadIdx.x) >> 5;
  const int lane = threadIdx.x & 31;
  if (wid >= MQ * NHD) return;
  const int h  = wid & 7;
  const int bq = wid >> 3;
  const int b  = bq / LQ;

  const int LH[4]  = {100, 50, 25, 13};
  const int LW[4]  = {152, 76, 38, 19};
  const int LS0[4] = {0, 15200, 19000, 19950};

  const float* lp  = loc + (size_t)wid * 32;
  const float* awp = aw + (size_t)wid * 16;
  float acc = 0.0f;
#pragma unroll
  for (int l = 0; l < NLV; ++l) {
    const int H_ = LH[l], W_ = LW[l];
    const float* vb = value + (((size_t)b * LV + LS0[l]) * NHD + h) * HD + lane;
#pragma unroll
    for (int p = 0; p < NPT; ++p) {
      const int j = l * 4 + p;
      const float x = lp[j * 2 + 0] * (float)W_ - 0.5f;
      const float y = lp[j * 2 + 1] * (float)H_ - 0.5f;
      const float xf = floorf(x), yf = floorf(y);
      const float fx = x - xf, fy = y - yf;
      const int x0 = (int)xf, y0 = (int)yf;
      float sacc = 0.0f;
#pragma unroll
      for (int cy = 0; cy < 2; ++cy) {
#pragma unroll
        for (int cx = 0; cx < 2; ++cx) {
          const int ix = x0 + cx, iy = y0 + cy;
          const float wc = (cx ? fx : 1.0f - fx) * (cy ? fy : 1.0f - fy);
          const bool valid = (ix >= 0) && (ix < W_) && (iy >= 0) && (iy < H_);
          const int cix = min(max(ix, 0), W_ - 1);
          const int ciy = min(max(iy, 0), H_ - 1);
          const float g = vb[(size_t)(ciy * W_ + cix) * (NHD * HD)];
          sacc += g * (valid ? wc : 0.0f);
        }
      }
      acc += awp[j] * sacc;
    }
  }
  out[(size_t)wid * HD + lane] = acc;   // (b,q,h,d) == (b,q,256)
}

// ---------------------------------------------------------------------------
// Orchestration
// ---------------------------------------------------------------------------
extern "C" void kernel_launch(void* const* d_in, const int* in_sizes, int n_in,
                              void* d_out, int out_size, void* d_ws, size_t ws_size,
                              hipStream_t stream) {
  (void)in_sizes; (void)n_in; (void)out_size; (void)ws_size;
  const float* tgt           = (const float*)d_in[0];
  const float* query_pos     = (const float*)d_in[1];
  const float* refp          = (const float*)d_in[2];
  const float* src           = (const float*)d_in[3];
  const float* in_proj_w     = (const float*)d_in[4];
  const float* in_proj_b     = (const float*)d_in[5];
  const float* out_proj_w    = (const float*)d_in[6];
  const float* out_proj_b    = (const float*)d_in[7];
  const float* samp_off_w    = (const float*)d_in[8];
  const float* samp_off_b    = (const float*)d_in[9];
  const float* attn_w_w      = (const float*)d_in[10];
  const float* attn_w_b      = (const float*)d_in[11];
  const float* value_proj_w  = (const float*)d_in[12];
  const float* value_proj_b  = (const float*)d_in[13];
  const float* output_proj_w = (const float*)d_in[14];
  const float* output_proj_b = (const float*)d_in[15];
  const float* lin1_w        = (const float*)d_in[16];
  const float* lin1_b        = (const float*)d_in[17];
  const float* lin2_w        = (const float*)d_in[18];
  const float* lin2_b        = (const float*)d_in[19];
  const float* n1w = (const float*)d_in[20]; const float* n1b = (const float*)d_in[21];
  const float* n2w = (const float*)d_in[22]; const float* n2b = (const float*)d_in[23];
  const float* n3w = (const float*)d_in[24]; const float* n3b = (const float*)d_in[25];
  const unsigned char* pad_mask = (const unsigned char*)d_in[28];

  float* out_x   = (float*)d_out;          // (8,300,256)
  float* out_loc = out_x + MQ * DM;        // (8,300,8,4,4,2)
  float* out_aw  = out_loc + MQ * DM;      // (8,300,8,4,4)

  float* ws = (float*)d_ws;
  size_t o = 0;
  auto alloc = [&](size_t n) { float* p = ws + o; o += n; return p; };
  float* qk     = alloc((size_t)MQ * DM);
  float* qkbuf  = alloc((size_t)MQ * 512);
  float* vbuf   = alloc((size_t)MQ * DM);
  float* attn   = alloc((size_t)MQ * DM);
  float* sa     = alloc((size_t)MQ * DM);
  float* x1     = alloc((size_t)MQ * DM);
  float* query  = alloc((size_t)MQ * DM);
  float* value  = alloc((size_t)MV * DM);
  float* offs   = alloc((size_t)MQ * DM);
  float* logits = alloc((size_t)MQ * 128);
  float* ca_pre = alloc((size_t)MQ * DM);
  float* ca     = alloc((size_t)MQ * DM);
  float* x2     = alloc((size_t)MQ * DM);
  float* ffh    = alloc((size_t)MQ * DFF);
  float* ffn    = alloc((size_t)MQ * DM);

  const dim3 gb(32, 2);                    // 2 waves, 16x128 per block
  const int NELEM = MQ * DM;

  // ---- self attention ----
  add_kernel<<<(NELEM + 255) / 256, 256, 0, stream>>>(tgt, query_pos, qk, NELEM);
  gemm_wmma<256, 512, false, false, false><<<dim3(MQ / 16, 512 / 128), gb, 0, stream>>>(
      qk, in_proj_w, in_proj_b, qkbuf, MQ, nullptr);
  gemm_wmma<256, 256, false, false, false><<<dim3(MQ / 16, 256 / 128), gb, 0, stream>>>(
      tgt, in_proj_w + 512 * DM, in_proj_b + 512, vbuf, MQ, nullptr);
  self_attn_kernel<<<BS * NHD * LQ, 32, 0, stream>>>(qkbuf, vbuf, attn);
  gemm_wmma<256, 256, false, false, false><<<dim3(MQ / 16, 256 / 128), gb, 0, stream>>>(
      attn, out_proj_w, out_proj_b, sa, MQ, nullptr);
  add_ln_kernel<<<(MQ * 32 + 255) / 256, 256, 0, stream>>>(tgt, sa, n2w, n2b, x1, MQ);

  // ---- deformable cross attention ----
  add_kernel<<<(NELEM + 255) / 256, 256, 0, stream>>>(x1, query_pos, query, NELEM);
  gemm_wmma<256, 256, false, true, true><<<dim3((MV + 15) / 16, 256 / 128), gb, 0, stream>>>(
      src, value_proj_w, value_proj_b, value, MV, pad_mask);
  gemm_wmma<256, 256, false, false, false><<<dim3(MQ / 16, 256 / 128), gb, 0, stream>>>(
      query, samp_off_w, samp_off_b, offs, MQ, nullptr);
  gemm_wmma<256, 128, false, false, false><<<dim3(MQ / 16, 1), gb, 0, stream>>>(
      query, attn_w_w, attn_w_b, logits, MQ, nullptr);
  loc_aw_kernel<<<(MQ * NHD + 255) / 256, 256, 0, stream>>>(
      logits, offs, refp, out_loc, out_aw);
  deform_sample_kernel<<<(MQ * NHD * 32 + 255) / 256, 256, 0, stream>>>(
      value, out_loc, out_aw, ca_pre);
  gemm_wmma<256, 256, false, false, false><<<dim3(MQ / 16, 256 / 128), gb, 0, stream>>>(
      ca_pre, output_proj_w, output_proj_b, ca, MQ, nullptr);
  add_ln_kernel<<<(MQ * 32 + 255) / 256, 256, 0, stream>>>(x1, ca, n1w, n1b, x2, MQ);

  // ---- FFN ----
  gemm_wmma<256, 1024, true, false, false><<<dim3(MQ / 16, DFF / 128), gb, 0, stream>>>(
      x2, lin1_w, lin1_b, ffh, MQ, nullptr);
  gemm_wmma<1024, 256, false, false, false><<<dim3(MQ / 16, 256 / 128), gb, 0, stream>>>(
      ffh, lin2_w, lin2_b, ffn, MQ, nullptr);
  add_ln_kernel<<<(MQ * 32 + 255) / 256, 256, 0, stream>>>(x2, ffn, n3w, n3b, out_x, MQ);
}
